// DAGExecutor_43499428774299
// MI455X (gfx1250) — matrix operand order repaired and checked by reading, not compiled, not run on hardware
//
#include <hip/hip_runtime.h>
#include <math.h>

typedef _Float16 v16h __attribute__((ext_vector_type(16)));
typedef _Float16 v8h  __attribute__((ext_vector_type(8)));
typedef float    v8f  __attribute__((ext_vector_type(8)));

#define WAVES_PER_BLOCK 2
#define TOKENS_PER_WAVE 32
#define N_INIT   9
#define TOTAL    17
#define DEPTH    8
#define K_TOT    72          // 9 nodes * 8 digits
#define E_STRIDE 80          // LDS row stride in halves (16B-aligned chunks)

__global__ __launch_bounds__(WAVES_PER_BLOCK * 32)
void dag_exec_kernel(const float* __restrict__ dl,   // (tok, 9, 8, 10)
                     const float* __restrict__ vs,   // (tok, 17)
                     const float* __restrict__ Om,   // (tok, 8, 17)
                     const float* __restrict__ Gm,   // (tok, 8)
                     float* __restrict__ out)        // (tok)
{
    __shared__ __align__(16) _Float16 eSm[WAVES_PER_BLOCK * TOKENS_PER_WAVE * E_STRIDE];
    __shared__ float magSm[WAVES_PER_BLOCK * TOKENS_PER_WAVE * TOTAL];

    const int lane = threadIdx.x & 31;
    const int wib  = threadIdx.x >> 5;
    const long tokenBase = ((long)blockIdx.x * WAVES_PER_BLOCK + wib) * TOKENS_PER_WAVE;

    _Float16* eW  = eSm  + wib * TOKENS_PER_WAVE * E_STRIDE;
    float*    magW = magSm + wib * TOKENS_PER_WAVE * TOTAL;

    // ---------- Phase A: softmax expected digits -> LDS (f16) ----------
    // 32 tokens * 72 (node,digit) groups; each lane handles 72 groups.
    {
        unsigned g = (unsigned)lane;
        for (int it = 0; it < 72; ++it, g += 32) {
            unsigned tt = g / 72u;            // token-in-wave
            unsigned k  = g - tt * 72u;       // n*8 + d
            const float2* p = (const float2*)(dl + (tokenBase + tt) * 720 + (long)k * 10);
            float2 a0 = p[0], a1 = p[1], a2 = p[2], a3 = p[3], a4 = p[4];
            float x[10] = {a0.x, a0.y, a1.x, a1.y, a2.x, a2.y, a3.x, a3.y, a4.x, a4.y};
            float m = x[0];
            #pragma unroll
            for (int v = 1; v < 10; ++v) m = fmaxf(m, x[v]);
            float den = 0.f, num = 0.f;
            #pragma unroll
            for (int v = 0; v < 10; ++v) {
                float e = __expf((x[v] - m) * 100.0f);   // 1/TEMPERATURE
                den += e;
                num += e * (float)v;
            }
            eW[tt * E_STRIDE + k] = (_Float16)(num / den);
        }
    }
    __syncthreads();

    // ---------- Phase B: WMMA  mag[t][n] = sum_d e[t][n*8+d] * 10^(3-d) ----------
    const int nCol   = lane & 15;
    const int laneHi = lane >> 4;
    const float p10[8] = {1e3f, 1e2f, 1e1f, 1.f, 1e-1f, 1e-2f, 1e-3f, 1e-4f};

    // Constant block-diagonal B fragments (32x16 f16 layout: lanes<16 K=0..7,16..23, lanes>=16 K=8..15,24..31)
    v16h bfrag[3];
    #pragma unroll
    for (int i = 0; i < 3; ++i) {
        #pragma unroll
        for (int j = 0; j < 16; ++j) {
            int K = 32 * i + (laneHi ? 8 : 0) + (j & 7) + ((j >> 3) << 4);
            float v = (K < K_TOT && (K >> 3) == nCol) ? p10[K & 7] : 0.0f;
            bfrag[i][j] = (_Float16)v;
        }
    }

    #pragma unroll
    for (int h = 0; h < 2; ++h) {               // two 16-token tiles
        v8f c = {};
        const _Float16* eRow = eW + (16 * h + nCol) * E_STRIDE;  // A row M = lane&15
        #pragma unroll
        for (int i = 0; i < 3; ++i) {           // K = 96 (72 valid, rest zero)
            int kLo = 32 * i + (laneHi ? 8 : 0);
            int kHi = kLo + 16;
            v8h lo = {}, hi = {};
            if (kLo < K_TOT) lo = *(const v8h*)(eRow + kLo);
            if (kHi < K_TOT) hi = *(const v8h*)(eRow + kHi);
            v16h a;
            #pragma unroll
            for (int j = 0; j < 8; ++j) { a[j] = lo[j]; a[8 + j] = hi[j]; }
            c = __builtin_amdgcn_wmma_f32_16x16x32_f16(false, a, false, bfrag[i],
                                                       (short)0, c, false, false);
        }
        // D layout: VGPR r -> M = r + 8*laneHi, N = lane&15
        #pragma unroll
        for (int r = 0; r < 8; ++r) {
            int M = r + (laneHi ? 8 : 0);
            float mval = fminf(fmaxf(c[r], 1e-8f), 1e6f);
            magW[(16 * h + M) * TOTAL + nCol] = mval;
        }
    }
    __syncthreads();

    // ---------- Phase D: per-token DAG recurrence (one token per lane) ----------
    const long token = tokenBase + lane;
    float mg[TOTAL], sg[TOTAL], lm[TOTAL];
    #pragma unroll
    for (int n = 0; n < N_INIT; ++n) {
        mg[n] = magW[lane * TOTAL + n];
        lm[n] = logf(fmaxf(mg[n], 1e-12f));
    }
    const float* vsp = vs + token * TOTAL;
    #pragma unroll
    for (int j = 0; j < TOTAL; ++j) sg[j] = vsp[j];

    const float* Op = Om + token * (DEPTH * TOTAL);
    const float* Gp = Gm + token * DEPTH;

    #pragma unroll
    for (int s = 0; s < DEPTH; ++s) {
        float Gs  = Gp[s];
        float om1 = 1.0f - Gs;
        float R = 0.f, P = 1.f;
        #pragma unroll
        for (int j = 0; j < N_INIT + s; ++j) {   // causal mask: j < 9+s
            float o = Op[s * TOTAL + j];
            float mixed = lm[j] * om1 + sg[j] * mg[j] * Gs;
            R += o * mixed;
            P *= fmaf(sg[j] * fabsf(o), 2.0f, 1.0f);
        }
        float lin = tanhf(R * 1e4f);
        float lgs = tanhf(P * 1e4f);
        float sn = Gs * lin + om1 * lgs;
        float Rc = fminf(fmaxf(R, -100.f), 100.f);
        float mn = Gs * fabsf(R) + om1 * expf(Rc);
        int idx = N_INIT + s;
        mg[idx] = mn;
        sg[idx] = sn;
        lm[idx] = logf(fmaxf(mn, 1e-12f));
    }
    out[token] = sg[TOTAL - 1] * mg[TOTAL - 1];
}

extern "C" void kernel_launch(void* const* d_in, const int* in_sizes, int n_in,
                              void* d_out, int out_size, void* d_ws, size_t ws_size,
                              hipStream_t stream) {
    (void)n_in; (void)out_size; (void)d_ws; (void)ws_size;
    const float* dl = (const float*)d_in[0];   // digit_logits
    const float* vs = (const float*)d_in[1];   // V_sign
    const float* Om = (const float*)d_in[2];   // O
    const float* Gm = (const float*)d_in[3];   // G
    float* out = (float*)d_out;

    int tokens = in_sizes[1] / TOTAL;                        // B*T = 16384
    int tokensPerBlock = WAVES_PER_BLOCK * TOKENS_PER_WAVE;  // 64
    int blocks = tokens / tokensPerBlock;                    // 256
    dag_exec_kernel<<<blocks, WAVES_PER_BLOCK * 32, 0, stream>>>(dl, vs, Om, Gm, out);
}
